// DAttentionDCN_7851200217271
// MI455X (gfx1250) — matrix operand order repaired and checked by reading, not compiled, not run on hardware
//
#include <hip/hip_runtime.h>
#include <hip/hip_bf16.h>

// ---------------------------------------------------------------------------
// Deformable attention (DCN-style) for MI455X / gfx1250 (wave32, WMMA).
// B=8, C=256, H=W=64, L=4096, A=8 heads, N=4 samples, D=32.
// GEMMs use v_wmma_f32_16x16x32_bf16; sampling is a VALU gather kernel.
// ---------------------------------------------------------------------------

#define Bsz 8
#define Cc  256
#define Hh  64
#define Ww  64
#define Ll  4096   // H*W
#define Ah  8
#define Ns  4

typedef __attribute__((ext_vector_type(16))) __bf16 bf16x16;
typedef __attribute__((ext_vector_type(8)))  __bf16 bf16x8;
typedef __attribute__((ext_vector_type(8)))  float  v8f;

__device__ __forceinline__ unsigned short f2bf(float f) {
    unsigned u = __float_as_uint(f);
    // round-to-nearest-even bf16
    unsigned r = (u + 0x7FFFu + ((u >> 16) & 1u)) >> 16;
    return (unsigned short)r;
}

// ---------------------------------------------------------------------------
// 1) x [B,C,L] fp32  ->  xb [B,L,C] bf16   (LDS tiled transpose, 32x32 tiles)
// ---------------------------------------------------------------------------
__global__ void __launch_bounds__(256)
k_transpose_bf16(const float* __restrict__ x, unsigned short* __restrict__ xb)
{
    __shared__ float tile[32][33];
    const int b  = blockIdx.z;
    const int c0 = blockIdx.x * 32;
    const int l0 = blockIdx.y * 32;
    const int tx = threadIdx.x;      // 0..31  (l within tile on read)
    const int ty = threadIdx.y;      // 0..7
    const float* src = x + ((size_t)b * Cc + c0) * Ll + l0;
#pragma unroll
    for (int i = 0; i < 32; i += 8)
        tile[ty + i][tx] = src[(size_t)(ty + i) * Ll + tx];
    __syncthreads();
    unsigned short* dst = xb + ((size_t)b * Ll + l0) * Cc + c0;
#pragma unroll
    for (int i = 0; i < 32; i += 8)
        dst[(size_t)(ty + i) * Cc + tx] = f2bf(tile[tx][ty + i]);
}

// ---------------------------------------------------------------------------
// 2) fp32 -> bf16 elementwise (weights)
// ---------------------------------------------------------------------------
__global__ void k_cvt_bf16(const float* __restrict__ in,
                           unsigned short* __restrict__ out, int n)
{
    int i = blockIdx.x * 256 + threadIdx.x;
    if (i < n) out[i] = f2bf(in[i]);
}

// ---------------------------------------------------------------------------
// 3) GEMM:  Out[b, l, o] (or Out[b, o, l] if TRANS) = sum_c A[b,l,c]*W[o,c] + bias[o]
//    A: [B, L, C] bf16 row-major (K contiguous), W: [O, C] bf16 (K contiguous).
//    One wave computes a 16(M=l) x 16*NT(N=o) D tile; K-loop over C in 32-steps.
//    A and B fragments both use contiguous k = 16*half + s (wave32).
// ---------------------------------------------------------------------------
template<int NT, bool TRANS>
__global__ void __launch_bounds__(256)
k_gemm_xw(const __bf16* __restrict__ A, const __bf16* __restrict__ W,
          const float* __restrict__ bias, float* __restrict__ Out, int O)
{
    const int wave = threadIdx.x >> 5;
    const int lane = threadIdx.x & 31;
    const int half = lane >> 4;       // K sub-range selector
    const int mr   = lane & 15;       // A row / B column within tile

    const int tilesN = O / (16 * NT);
    const int tilesM = Ll / 16;
    const int perB   = tilesM * tilesN;

    int wid = blockIdx.x * 8 + wave;
    int b   = wid / perB;
    if (b >= Bsz) return;
    int rem = wid - b * perB;
    int mt  = rem / tilesN;
    int nt  = rem - mt * tilesN;
    const int n0 = nt * 16 * NT;

    const __bf16* Ab = A + (size_t)b * Ll * Cc + (size_t)(mt * 16 + mr) * Cc;

    v8f acc[NT];
#pragma unroll
    for (int t = 0; t < NT; t++) {
        v8f z = {0.f, 0.f, 0.f, 0.f, 0.f, 0.f, 0.f, 0.f};
        acc[t] = z;
    }

#pragma unroll
    for (int k0 = 0; k0 < Cc; k0 += 32) {
        if (k0 + 32 < Cc)
            __builtin_prefetch(Ab + k0 + 32, 0, 1);   // global_prefetch next A chunk

        // A fragment (16-bit A 16x32): lanes 0-15 -> K {k0+0..15},
        // lanes 16-31 -> K {k0+16..31}
        const __bf16* pa = Ab + k0 + half * 16;
        bf16x8 alo = *(const bf16x8*)(pa);
        bf16x8 ahi = *(const bf16x8*)(pa + 8);
        bf16x16 afrag = __builtin_shufflevector(alo, ahi,
                             0, 1, 2, 3, 4, 5, 6, 7, 8, 9, 10, 11, 12, 13, 14, 15);

#pragma unroll
        for (int t = 0; t < NT; t++) {
            // B fragment (32x16): lane -> column n, half-wave selects K 0..15 / 16..31
            const __bf16* pb = W + (size_t)(n0 + t * 16 + mr) * Cc + k0 + half * 16;
            bf16x16 bfrag = *(const bf16x16*)pb;
            acc[t] = __builtin_amdgcn_wmma_f32_16x16x32_bf16(
                         false, afrag, false, bfrag, (short)0, acc[t], false, false);
        }
    }

    // Epilogue: D layout — VGPR r holds (M = r + 8*half, N = lane&15)
#pragma unroll
    for (int t = 0; t < NT; t++) {
        const int ncol = n0 + t * 16 + mr;
        const float bv = bias[ncol];
#pragma unroll
        for (int r = 0; r < 8; r++) {
            const int mrow = mt * 16 + r + half * 8;
            const float val = acc[t][r] + bv;
            if (!TRANS)
                Out[((size_t)b * Ll + mrow) * O + ncol] = val;
            else
                Out[((size_t)b * O + ncol) * Ll + mrow] = val;
        }
    }
}

// ---------------------------------------------------------------------------
// 4) Fused softmax + offsets->positions + bilinear sample + head aggregation.
//    One wave per (b,l); lane = channel-in-head. Per-(head,sample) weight and
//    position computed by lane j=(a*4+s), broadcast via LDS.
//    attL [B,L,32] fp32, offP [B,L,64] fp32, vP [B,L,256] fp32
//    -> aggb [B,L,256] bf16 (input to the final GEMM)
// ---------------------------------------------------------------------------
__global__ void __launch_bounds__(256)
k_dcn_sample(const float* __restrict__ attL, const float* __restrict__ offP,
             const float* __restrict__ vP, unsigned short* __restrict__ aggb)
{
    __shared__ float s_w[8][32];
    __shared__ float s_y[8][32];
    __shared__ float s_x[8][32];

    const int wave = threadIdx.x >> 5;
    const int lane = threadIdx.x & 31;
    const size_t pos = (size_t)blockIdx.x * 8 + wave;    // b*L + l
    const int b  = (int)(pos >> 12);
    const int l  = (int)(pos & (Ll - 1));
    const int iy = l >> 6;
    const int ix = l & 63;

    // --- softmax over N=4 within head: lane j -> (a=j/4, n=j%4), groups of 4 lanes
    float logit = attL[pos * 32 + lane];
    float m = logit;
    m = fmaxf(m, __shfl_xor(m, 1, 32));
    m = fmaxf(m, __shfl_xor(m, 2, 32));
    float e = __expf(logit - m);
    float ssum = e;
    ssum += __shfl_xor(ssum, 1, 32);
    ssum += __shfl_xor(ssum, 2, 32);
    s_w[wave][lane] = e / ssum;

    // --- sampling position (align_corners=True mapping)
    const float refy = ((iy + 0.5f) / (float)(Hh - 1)) * 2.f - 1.f;
    const float refx = ((ix + 0.5f) / (float)(Ww - 1)) * 2.f - 1.f;
    const float oy = offP[pos * 64 + lane * 2 + 0];
    const float ox = offP[pos * 64 + lane * 2 + 1];
    s_y[wave][lane] = (refy + oy + 1.f) * 0.5f * (float)(Hh - 1);
    s_x[wave][lane] = (refx + ox + 1.f) * 0.5f * (float)(Ww - 1);
    __syncthreads();

    const float* vb = vP + (size_t)b * Ll * Cc;

#pragma unroll
    for (int a = 0; a < Ah; a++) {
        float agg = 0.f;
#pragma unroll
        for (int s = 0; s < Ns; s++) {
            const int j = a * 4 + s;
            const float w = s_w[wave][j];
            const float y = s_y[wave][j];
            const float x = s_x[wave][j];
            const float y0 = floorf(y), x0 = floorf(x);
#pragma unroll
            for (int dy = 0; dy < 2; dy++) {
#pragma unroll
                for (int dx = 0; dx < 2; dx++) {
                    const float yi = y0 + (float)dy;
                    const float xi = x0 + (float)dx;
                    const float cw = (1.f - fabsf(y - yi)) * (1.f - fabsf(x - xi));
                    const bool valid = (yi >= 0.f) && (yi < (float)Hh) &&
                                       (xi >= 0.f) && (xi < (float)Ww);
                    int yc = (int)yi; yc = yc < 0 ? 0 : (yc > Hh - 1 ? Hh - 1 : yc);
                    int xc = (int)xi; xc = xc < 0 ? 0 : (xc > Ww - 1 ? Ww - 1 : xc);
                    const int lidx = yc * Ww + xc;
                    // lane = channel within head -> 32 contiguous floats per corner
                    const float val = vb[(size_t)lidx * Cc + a * 32 + lane];
                    agg += valid ? (w * cw * val) : 0.f;
                }
            }
        }
        aggb[pos * Cc + a * 32 + lane] = f2bf(agg);
    }
}

// ---------------------------------------------------------------------------
extern "C" void kernel_launch(void* const* d_in, const int* in_sizes, int n_in,
                              void* d_out, int out_size, void* d_ws, size_t ws_size,
                              hipStream_t stream)
{
    const float* x     = (const float*)d_in[0];   // [B,C,H,W]
    const float* w_att = (const float*)d_in[1];   // [32,256]
    const float* b_att = (const float*)d_in[2];   // [32]
    const float* w_off = (const float*)d_in[3];   // [64,256]
    const float* b_off = (const float*)d_in[4];   // [64]
    const float* w_v   = (const float*)d_in[5];   // [256,256]
    const float* b_v   = (const float*)d_in[6];   // [256]
    const float* w_out = (const float*)d_in[7];   // [256,256]
    const float* b_out = (const float*)d_in[8];   // [256]
    float* y = (float*)d_out;                     // [B,C,H,W]

    // ---- workspace layout (bytes, 256-aligned) ----
    char* ws = (char*)d_ws;
    size_t off = 0;
    auto alloc = [&](size_t bytes) { void* p = ws + off;
                                     off = (off + bytes + 255) & ~(size_t)255;
                                     return p; };
    unsigned short* xb    = (unsigned short*)alloc((size_t)Bsz * Ll * Cc * 2); // 16 MB
    unsigned short* wab   = (unsigned short*)alloc((size_t)32  * Cc * 2);
    unsigned short* wob   = (unsigned short*)alloc((size_t)64  * Cc * 2);
    unsigned short* wvb   = (unsigned short*)alloc((size_t)Cc  * Cc * 2);
    unsigned short* woutb = (unsigned short*)alloc((size_t)Cc  * Cc * 2);
    float*          attL  = (float*)alloc((size_t)Bsz * Ll * 32 * 4);          //  4 MB
    float*          offP  = (float*)alloc((size_t)Bsz * Ll * 64 * 4);          //  8 MB
    float*          vP    = (float*)alloc((size_t)Bsz * Ll * Cc * 4);          // 32 MB
    unsigned short* aggb  = (unsigned short*)alloc((size_t)Bsz * Ll * Cc * 2); // 16 MB
    (void)ws_size; (void)in_sizes; (void)n_in; (void)out_size;

    // 1) transpose + convert x -> [B,L,C] bf16
    k_transpose_bf16<<<dim3(Cc / 32, Ll / 32, Bsz), dim3(32, 8), 0, stream>>>(x, xb);

    // 2) weights -> bf16
    k_cvt_bf16<<<(32  * Cc + 255) / 256, 256, 0, stream>>>(w_att, wab,   32 * Cc);
    k_cvt_bf16<<<(64  * Cc + 255) / 256, 256, 0, stream>>>(w_off, wob,   64 * Cc);
    k_cvt_bf16<<<(Cc  * Cc + 255) / 256, 256, 0, stream>>>(w_v,   wvb,   Cc * Cc);
    k_cvt_bf16<<<(Cc  * Cc + 255) / 256, 256, 0, stream>>>(w_out, woutb, Cc * Cc);

    // 3) projections (WMMA GEMMs). blocks = B * (L/16) * tilesN / 8 waves
    {   // attn logits: O=32, NT=2 -> tilesN=1
        int blocks = Bsz * (Ll / 16) * 1 / 8;
        k_gemm_xw<2, false><<<blocks, 256, 0, stream>>>(
            (const __bf16*)xb, (const __bf16*)wab, b_att, attL, 32);
    }
    {   // offsets: O=64, NT=4 -> tilesN=1
        int blocks = Bsz * (Ll / 16) * 1 / 8;
        k_gemm_xw<4, false><<<blocks, 256, 0, stream>>>(
            (const __bf16*)xb, (const __bf16*)wob, b_off, offP, 64);
    }
    {   // values: O=256, NT=4 -> tilesN=4
        int blocks = Bsz * (Ll / 16) * 4 / 8;
        k_gemm_xw<4, false><<<blocks, 256, 0, stream>>>(
            (const __bf16*)xb, (const __bf16*)wvb, b_v, vP, Cc);
    }

    // 4) fused softmax + deformable bilinear sampling + aggregation
    k_dcn_sample<<<(Bsz * Ll) / 8, 256, 0, stream>>>(attL, offP, vP, aggb);

    // 5) output projection, transposed store -> y [B,C,H,W]
    {
        int blocks = Bsz * (Ll / 16) * 4 / 8;
        k_gemm_xw<4, true><<<blocks, 256, 0, stream>>>(
            (const __bf16*)aggb, (const __bf16*)woutb, b_out, y, Cc);
    }
}
